// Mol2Pair_50044958933069
// MI455X (gfx1250) — matrix operand order repaired and checked by reading, not compiled, not run on hardware
//
#include <hip/hip_runtime.h>

// ---------------------------------------------------------------------------
// MI455X / gfx1250 implementation. wave32, WMMA bf16 16x16x32 (f32 accumulate).
// ---------------------------------------------------------------------------

typedef __attribute__((ext_vector_type(16))) __bf16 v16bf;
typedef __attribute__((ext_vector_type(8)))  __bf16 v8bf;
typedef __attribute__((ext_vector_type(4)))  __bf16 v4bf;
typedef __attribute__((ext_vector_type(8)))  float  v8f;
typedef __attribute__((ext_vector_type(4)))  float  v4f;

#define DEV static __device__ __forceinline__

DEV float sigmoidf_(float x){ return 1.0f/(1.0f + __expf(-x)); }

DEV v16bf frag_from2(const __bf16* p0, const __bf16* p1){
  union { v16bf v; v8bf h[2]; } u;
  u.h[0] = *(const v8bf*)p0;
  u.h[1] = *(const v8bf*)p1;
  return u.v;
}

// A fragment (16x32 bf16) from row-major [M][K] (stride in elements).
// Lane l<16: row l, K chunks [k0..k0+7] and [k0+16..k0+23];
// lane l>=16: row l-16, chunks [k0+8..15] and [k0+24..31].
DEV v16bf load_afrag(const __bf16* base, int stride, int k0, int lane){
  int row  = lane & 15;
  int koff = k0 + ((lane >> 4) << 3);
  const __bf16* p = base + (size_t)row*stride + koff;
  return frag_from2(p, p + 16);
}

// B fragment (32x16 bf16) from BT layout [N][K] row-major: two 16B loads
// per lane (lane l<16: col n0+l, K chunks [k0..k0+7] and [k0+16..k0+23];
// l>=16: chunks [k0+8..15] and [k0+24..31]).
DEV v16bf load_bfrag(const __bf16* bt, int stride, int n0, int k0, int lane){
  int n    = n0 + (lane & 15);
  int koff = k0 + ((lane >> 4) << 3);
  const __bf16* p = bt + (size_t)n*stride + koff;
  return frag_from2(p, p + 16);
}

DEV v8f wmma_bf16(v16bf a, v16bf b, v8f c){
  return __builtin_amdgcn_wmma_f32_16x16x32_bf16(false, a, false, b,
                                                 (short)0, c, false, false);
}

// ---------------------------------------------------------------------------
// K1: row LayerNorm over 128 channels, f32 -> bf16. One wave per row.
// ---------------------------------------------------------------------------
__global__ void __launch_bounds__(256) k_ln(const float* __restrict__ x,
                                            const float* __restrict__ g,
                                            const float* __restrict__ bta,
                                            __bf16* __restrict__ y, int nrows)
{
  int wave = threadIdx.x >> 5, lane = threadIdx.x & 31;
  size_t row = (size_t)blockIdx.x*8 + wave;
  if (row >= (size_t)nrows) return;
  v4f v = *(const v4f*)(x + row*128 + lane*4);
  float s = v[0]+v[1]+v[2]+v[3];
  float q = v[0]*v[0]+v[1]*v[1]+v[2]*v[2]+v[3]*v[3];
  #pragma unroll
  for (int m = 16; m >= 1; m >>= 1){ s += __shfl_xor(s, m, 32); q += __shfl_xor(q, m, 32); }
  float mu = s*(1.0f/128.0f);
  float rs = rsqrtf(q*(1.0f/128.0f) - mu*mu + 1e-5f);
  v4f gg = *(const v4f*)(g   + lane*4);
  v4f bb = *(const v4f*)(bta + lane*4);
  v4bf o;
  #pragma unroll
  for (int i=0;i<4;++i) o[i] = (__bf16)((v[i]-mu)*rs*gg[i] + bb[i]);
  *(v4bf*)(y + row*128 + lane*4) = o;
}

// ---------------------------------------------------------------------------
// K0: pack six 128x128 f32 weights into bf16 transposed WT[n][k] = W[k][n].
// ---------------------------------------------------------------------------
__global__ void __launch_bounds__(256) k_packw(const float* w0,const float* w1,
                                               const float* w2,const float* w3,
                                               const float* w4,const float* w5,
                                               __bf16* out)
{
  const float* srcs[6] = {w0,w1,w2,w3,w4,w5};
  const float* W = srcs[blockIdx.x];
  __bf16* O = out + (size_t)blockIdx.x*16384;
  for (int it=0; it<64; ++it){
    int e = threadIdx.x + it*256;
    int n = e >> 7, k = e & 127;
    O[e] = (__bf16)W[k*128 + n];
  }
}

// ---------------------------------------------------------------------------
// K2: gated projection. out[b][c][p] = sigmoid(x@wG + bG)*(x@wV + bV)*mask,
// stored channel-major (contiguous packed-bf16 stores per lane).
// 128 rows per workgroup; 8 waves, each 16 rows x 128 cols x 2 GEMMs.
// ---------------------------------------------------------------------------
__global__ void __launch_bounds__(256) k_gate(const __bf16* __restrict__ src,
                       const __bf16* __restrict__ wVT, const __bf16* __restrict__ wGT,
                       const float* __restrict__ bV, const float* __restrict__ bG,
                       const float* __restrict__ mask, __bf16* __restrict__ out,
                       int plane)
{
  __shared__ __bf16 As[128][136];
  __shared__ float  MS[128];
  size_t gr0 = (size_t)blockIdx.x * 128;
  int b  = (int)(gr0 / (size_t)plane);
  int p0 = (int)(gr0 % (size_t)plane);
  int t = threadIdx.x;
  #pragma unroll
  for (int it=0; it<8; ++it){
    int cidx = t + it*256;           // 2048 chunks of 8 bf16
    int row  = cidx >> 4;
    int k0   = (cidx & 15) << 3;
    *(v8bf*)&As[row][k0] = *(const v8bf*)(src + (gr0 + row)*128 + k0);
  }
  if (t < 128) MS[t] = mask ? mask[gr0 + t] : 1.0f;
  __syncthreads();

  int w = t >> 5, lane = t & 31;
  int m0 = w*16;
  v8f zero = {0.f,0.f,0.f,0.f,0.f,0.f,0.f,0.f};
  v8f accG[8], accV[8];
  #pragma unroll
  for (int nb=0; nb<8; ++nb){ accG[nb]=zero; accV[nb]=zero; }

  #pragma unroll
  for (int ks=0; ks<4; ++ks){        // K = 128
    int k0 = ks*32;
    v16bf a = load_afrag(&As[m0][0], 136, k0, lane);
    #pragma unroll
    for (int nb=0; nb<8; ++nb){
      v16bf bg = load_bfrag(wGT, 128, nb*16, k0, lane);
      accG[nb] = wmma_bf16(a, bg, accG[nb]);
      v16bf bv = load_bfrag(wVT, 128, nb*16, k0, lane);
      accV[nb] = wmma_bf16(a, bv, accV[nb]);
    }
  }
  int hi = lane >> 4, nloc = lane & 15;
  #pragma unroll
  for (int nb=0; nb<8; ++nb){
    int c = nb*16 + nloc;
    float gb = bG[c], vb = bV[c];
    union { v8bf v; __bf16 e[8]; } pk;
    #pragma unroll
    for (int r=0; r<8; ++r){
      int mrow = m0 + 8*hi + r;
      float gv = sigmoidf_(accG[nb][r] + gb);
      float vv = accV[nb][r] + vb;
      pk.e[r] = (__bf16)(gv * vv * MS[mrow]);
    }
    __bf16* op = out + (size_t)(b*128 + c)*(size_t)plane + p0 + m0 + 8*hi;
    *(v8bf*)op = pk.v;
  }
}

// ---------------------------------------------------------------------------
// K5: per-(b,c) triangle GEMMs, computed transposed:
//   BS[j][i] = sum_k ab1[k,j]*ppg[i,k] + sum_k mpg[j,k]*ab2[i,k]
// ab1 plane is transposed through LDS; ppg/ab2/mpg are used in natural layout
// (contiguous A/B fragment loads). One WG per (b, c, j-half of 64).
// ---------------------------------------------------------------------------
__global__ void __launch_bounds__(256) k_block(const __bf16* __restrict__ AB1,
                        const __bf16* __restrict__ AB2,
                        const __bf16* __restrict__ PPG,
                        const __bf16* __restrict__ MPG,
                        float* __restrict__ BS)
{
  __shared__ __bf16 A1s[64][264];   // ab1 transposed: [j'][k], K=256 (+8 pad)
  __shared__ __bf16 A2s[64][136];   // mpg natural:    [j'][k], K=128 (+8 pad)
  int wg = blockIdx.x;
  int b  = wg >> 8;
  int rr = wg & 255;
  int c  = rr >> 1;
  int jb = (rr & 1) * 64;
  size_t pl = (size_t)(b*128 + c);
  const __bf16* ab1p = AB1 + pl*32768;
  const __bf16* ab2p = AB2 + pl*32768;
  const __bf16* ppgp = PPG + pl*65536;
  const __bf16* mpgp = MPG + pl*16384;
  float* bsp = BS + pl*32768;
  int t = threadIdx.x;

  #pragma unroll
  for (int it=0; it<8; ++it){       // 256x64 bf16, transpose into LDS
    int cidx = t + it*256;
    int i  = cidx >> 3;             // 0..255 (contraction index)
    int j0 = (cidx & 7) << 3;
    v8bf d = *(const v8bf*)(ab1p + (size_t)i*128 + jb + j0);
    #pragma unroll
    for (int e=0;e<8;++e) A1s[j0+e][i] = d[e];
  }
  #pragma unroll
  for (int it=0; it<4; ++it){       // 64x128 bf16 straight copy
    int cidx = t + it*256;
    int jm = cidx >> 4;
    int k0 = (cidx & 15) << 3;
    *(v8bf*)&A2s[jm][k0] = *(const v8bf*)(mpgp + (size_t)(jb+jm)*128 + k0);
  }
  __syncthreads();

  int w = t >> 5, lane = t & 31;
  int i0 = w*32;                    // 8 waves x 32 i-cols = 256
  v8f zero = {0.f,0.f,0.f,0.f,0.f,0.f,0.f,0.f};
  v8f acc[4][2];
  #pragma unroll
  for (int mb=0; mb<4; ++mb){ acc[mb][0]=zero; acc[mb][1]=zero; }

  #pragma unroll 2
  for (int ks=0; ks<8; ++ks){       // GEMM1: K = 256 (pocket index), 8 steps
    int k0 = ks*32;
    v16bf b0 = load_bfrag(ppgp, 256, i0,    k0, lane);
    v16bf b1 = load_bfrag(ppgp, 256, i0+16, k0, lane);
    #pragma unroll
    for (int mb=0; mb<4; ++mb){
      v16bf a = load_afrag(&A1s[mb*16][0], 264, k0, lane);
      acc[mb][0] = wmma_bf16(a, b0, acc[mb][0]);
      acc[mb][1] = wmma_bf16(a, b1, acc[mb][1]);
    }
  }
  #pragma unroll
  for (int ks=0; ks<4; ++ks){       // GEMM2: K = 128 (mol index), 4 steps
    int k0 = ks*32;
    v16bf b0 = load_bfrag(ab2p, 128, i0,    k0, lane);
    v16bf b1 = load_bfrag(ab2p, 128, i0+16, k0, lane);
    #pragma unroll
    for (int mb=0; mb<4; ++mb){
      v16bf a = load_afrag(&A2s[mb*16][0], 136, k0, lane);
      acc[mb][0] = wmma_bf16(a, b0, acc[mb][0]);
      acc[mb][1] = wmma_bf16(a, b1, acc[mb][1]);
    }
  }
  int hi = lane >> 4, nloc = lane & 15;
  #pragma unroll
  for (int mb=0; mb<4; ++mb)
    #pragma unroll
    for (int nb=0; nb<2; ++nb)
      #pragma unroll
      for (int r=0; r<8; ++r){
        int jm = jb + mb*16 + 8*hi + r;
        int i  = i0 + nb*16 + nloc;
        bsp[(size_t)jm*256 + i] = acc[mb][nb][r];
      }
}

// ---------------------------------------------------------------------------
// K7: per-(b, j, i-quarter of 64): gather blocksum^T slice, LN over channels,
// out = sigmoid(zn@egw+egb) * (h@asw+asb) * mask, staged through LDS for
// coalesced f32 stores. 8 waves = 4 row-strips x 2 col-halves, 2 GEMMs each.
// ---------------------------------------------------------------------------
__global__ void __launch_bounds__(256) k_final(const float* __restrict__ BS,
                        const __bf16* __restrict__ ZN,
                        const __bf16* __restrict__ EGWT, const __bf16* __restrict__ ASWT,
                        const float* __restrict__ egb, const float* __restrict__ asb,
                        const float* __restrict__ lnhg, const float* __restrict__ lnhb,
                        const float* __restrict__ mask, float* __restrict__ out)
{
  __shared__ __align__(16) float Sraw[128*68];  // S[c][i'] then reused as O[i'][c]
  __shared__ __bf16 H[64][136];
  __shared__ float MS[64], LG[128], LB[128];
  int wg = blockIdx.x;
  int b  = wg >> 9;
  int rr = wg & 511;
  int j  = rr >> 2;
  int ib = (rr & 3) * 64;
  int t  = threadIdx.x;

  const float* bsb = BS + (size_t)b*128*32768 + (size_t)j*256 + ib;
  {
    int cc = t >> 1, half = (t & 1)*32;
    const float* srcr = bsb + (size_t)cc*32768 + half;
    float* dst = Sraw + cc*68 + half;
    #pragma unroll
    for (int q=0;q<8;++q) *(v4f*)(dst + q*4) = *(const v4f*)(srcr + q*4);
  }
  if (t < 64)  MS[t] = mask[((size_t)(b*256 + ib + t))*128 + j];
  if (t < 128){ LG[t] = lnhg[t]; LB[t] = lnhb[t]; }
  __syncthreads();

  if (t < 64){                       // LN over 128 channels per column i'
    float s = 0.f, q = 0.f;
    for (int c=0;c<128;++c){ float x = Sraw[c*68 + t]; s += x; q += x*x; }
    float mu = s*(1.0f/128.0f);
    float rs = rsqrtf(q*(1.0f/128.0f) - mu*mu + 1e-5f);
    for (int c=0;c<128;++c)
      H[t][c] = (__bf16)((Sraw[c*68 + t] - mu)*rs*LG[c] + LB[c]);
  }
  __syncthreads();

  int w = t >> 5, lane = t & 31;
  int m0 = (w & 3)*16, nc0 = (w >> 2)*64;
  v8f zero = {0.f,0.f,0.f,0.f,0.f,0.f,0.f,0.f};
  v8f accE[4], accA[4];
  #pragma unroll
  for (int nb=0; nb<4; ++nb){ accE[nb]=zero; accA[nb]=zero; }

  const __bf16* znb = ZN + (((size_t)(b*256 + ib))*128 + j)*128;
  #pragma unroll
  for (int ks=0; ks<4; ++ks){        // K = 128
    int k0 = ks*32;
    v16bf ah = load_afrag(&H[m0][0], 136, k0, lane);
    int zrow = m0 + (lane & 15);
    int koff = k0 + ((lane >> 4) << 3);
    const __bf16* zp = znb + (size_t)zrow*16384 + koff;  // zn row stride 128*128
    v16bf az = frag_from2(zp, zp + 16);
    #pragma unroll
    for (int nb=0; nb<4; ++nb){
      v16bf be = load_bfrag(EGWT, 128, nc0 + nb*16, k0, lane);
      accE[nb] = wmma_bf16(az, be, accE[nb]);
      v16bf ba = load_bfrag(ASWT, 128, nc0 + nb*16, k0, lane);
      accA[nb] = wmma_bf16(ah, ba, accA[nb]);
    }
  }
  int hi = lane >> 4, nloc = lane & 15;
  float* O = Sraw;                   // S is dead; reuse as output tile [i'][c]
  #pragma unroll
  for (int nb=0; nb<4; ++nb){
    int c = nc0 + nb*16 + nloc;
    float eb = egb[c], ab = asb[c];
    #pragma unroll
    for (int r=0; r<8; ++r){
      int il = m0 + 8*hi + r;
      O[il*132 + c] = sigmoidf_(accE[nb][r] + eb) * (accA[nb][r] + ab) * MS[il];
    }
  }
  __syncthreads();
  {
    int row = t >> 2, qr = (t & 3)*32;
    float* orow = out + (((size_t)(b*256 + ib + row))*128 + j)*128 + qr;
    const float* srcp = O + row*132 + qr;
    #pragma unroll
    for (int q=0;q<8;++q) *(v4f*)(orow + q*4) = *(const v4f*)(srcp + q*4);
  }
}

// ---------------------------------------------------------------------------
// Host launcher. ws layout (~268.7 MB):
//  ZN 32M | AB1 32M | AB2 32M | PPG 64M | MPG 16M | PPn/BS 64M | MPn 16M | WT
// ---------------------------------------------------------------------------
extern "C" void kernel_launch(void* const* d_in, const int* in_sizes, int n_in,
                              void* d_out, int out_size, void* d_ws, size_t ws_size,
                              hipStream_t stream)
{
  (void)in_sizes; (void)n_in; (void)out_size; (void)ws_size;
  const float* z    = (const float*)d_in[0];
  const float* zm   = (const float*)d_in[1];
  const float* pp   = (const float*)d_in[2];
  const float* mp   = (const float*)d_in[3];
  const float* lng  = (const float*)d_in[4];
  const float* lnb  = (const float*)d_in[5];
  const float* lnhg = (const float*)d_in[6];
  const float* lnhb = (const float*)d_in[7];
  const float* gw1  = (const float*)d_in[8];
  const float* gb1  = (const float*)d_in[9];
  const float* gw2  = (const float*)d_in[10];
  const float* gb2  = (const float*)d_in[11];
  const float* w1   = (const float*)d_in[12];
  const float* b1   = (const float*)d_in[13];
  const float* w2   = (const float*)d_in[14];
  const float* b2   = (const float*)d_in[15];
  const float* egw  = (const float*)d_in[16];
  const float* egb  = (const float*)d_in[17];
  const float* asw  = (const float*)d_in[18];
  const float* asb  = (const float*)d_in[19];
  float* outp = (float*)d_out;

  char* W = (char*)d_ws;
  size_t off = 0;
  __bf16* ZN  = (__bf16*)(W + off); off += (size_t)33554432;
  __bf16* AB1 = (__bf16*)(W + off); off += (size_t)33554432;
  __bf16* AB2 = (__bf16*)(W + off); off += (size_t)33554432;
  __bf16* PPG = (__bf16*)(W + off); off += (size_t)67108864;
  __bf16* MPG = (__bf16*)(W + off); off += (size_t)16777216;
  __bf16* PPn = (__bf16*)(W + off);
  float*  BS  = (float* )(W + off); off += (size_t)67108864; // BS reuses PPn
  __bf16* MPn = (__bf16*)(W + off); off += (size_t)16777216;
  __bf16* WT  = (__bf16*)(W + off); off += (size_t)196608;

  __bf16* w1T  = WT + 0*16384;
  __bf16* gw1T = WT + 1*16384;
  __bf16* w2T  = WT + 2*16384;
  __bf16* gw2T = WT + 3*16384;
  __bf16* egwT = WT + 4*16384;
  __bf16* aswT = WT + 5*16384;

  k_packw<<<6, 256, 0, stream>>>(w1, gw1, w2, gw2, egw, asw, WT);
  k_ln<<<131072/8, 256, 0, stream>>>(z,  lng, lnb, ZN,  131072);
  k_ln<<<262144/8, 256, 0, stream>>>(pp, lng, lnb, PPn, 262144);
  k_ln<<< 65536/8, 256, 0, stream>>>(mp, lng, lnb, MPn,  65536);
  k_gate<<<1024, 256, 0, stream>>>(ZN,  w1T, gw1T, b1, gb1, zm,      AB1, 32768);
  k_gate<<<1024, 256, 0, stream>>>(ZN,  w2T, gw2T, b2, gb2, zm,      AB2, 32768);
  k_gate<<<2048, 256, 0, stream>>>(PPn, w2T, gw2T, b2, gb2, nullptr, PPG, 65536);
  k_gate<<< 512, 256, 0, stream>>>(MPn, w1T, gw1T, b1, gb1, nullptr, MPG, 16384);
  k_block<<<1024, 256, 0, stream>>>(AB1, AB2, PPG, MPG, BS);
  k_final<<<2048, 256, 0, stream>>>(BS, ZN, egwT, aswT, egb, asb, lnhg, lnhb, zm, outp);
}
